// RHGNN_77129022701793
// MI455X (gfx1250) — matrix operand order repaired and checked by reading, not compiled, not run on hardware
//
#include <hip/hip_runtime.h>
#include <hip/hip_bf16.h>
#include <stdint.h>

// Problem constants (match reference)
#define NN    50000
#define EE    500000
#define DD    128
#define HH    4
#define HIDD  32
#define HD    128     // H*HID
#define RHIDD 32
#define LL    2
#define RR    2
#define SLOPEF 0.2f

typedef __attribute__((ext_vector_type(16))) __bf16 v16bf;
typedef __attribute__((ext_vector_type(8)))  __bf16 v8bf;
typedef __attribute__((ext_vector_type(8)))  float  v8f;
typedef __attribute__((ext_vector_type(4)))  unsigned int u32x4;
typedef __attribute__((ext_vector_type(8)))  int i32x8;
typedef __attribute__((ext_vector_type(4)))  int i32x4;

#if __has_builtin(__builtin_amdgcn_tensor_load_to_lds)
#define USE_TDM 1
#endif

__device__ __forceinline__ float lrelu(float v) { return v >= 0.f ? v : SLOPEF * v; }

// Monotone float -> uint encoding so atomicMax(uint) == float max
__device__ __forceinline__ unsigned fenc(float f) {
    unsigned u = __float_as_uint(f);
    return (u & 0x80000000u) ? ~u : (u | 0x80000000u);
}
__device__ __forceinline__ float fdec(unsigned u) {
    unsigned b = (u & 0x80000000u) ? (u ^ 0x80000000u) : ~u;
    return __uint_as_float(b);
}

__device__ __forceinline__ __bf16 f2bf(float f) {
    unsigned u = __float_as_uint(f);
    unsigned short r = (unsigned short)((u + 0x7fffu + ((u >> 16) & 1u)) >> 16);
    return __builtin_bit_cast(__bf16, r);
}

#ifdef USE_TDM
// Issue one TDM descriptor copying n8 contiguous 8-byte elements from global
// to LDS. D# layout per CDNA5 ISA 8.3/8.4: group0 = {count, lds_addr,
// global_addr[31:0], global_addr[56:32] | type=2}; group1 packs data_size=8B,
// tensor_dim0 = tile_dim0 = stride = n8, tensor_dim1 = tile_dim1 = 1.
__device__ __forceinline__ void tdm_load_lds(const void* gptr, void* ldsptr, unsigned n8)
{
    unsigned long long ga = (unsigned long long)(uintptr_t)gptr;
    unsigned lds = (unsigned)(uintptr_t)ldsptr;   // low 32 bits = LDS offset
    u32x4 g0;
    g0[0] = 1u;                                            // count=1, no gather
    g0[1] = lds;                                           // lds_addr
    g0[2] = (unsigned)(ga & 0xffffffffu);                  // global_addr[31:0]
    g0[3] = (unsigned)((ga >> 32) & 0x01ffffffu) | (2u << 30); // addr[56:32] | type=2
    i32x8 g1;
    g1[0] = (int)(3u << 16);                               // data_size = 8 bytes
    g1[1] = (int)((n8 & 0xffffu) << 16);                   // tensor_dim0[15:0]
    g1[2] = (int)(((n8 >> 16) & 0xffffu) | (1u << 16));    // tensor_dim0[31:16] | tensor_dim1=1
    g1[3] = (int)((n8 & 0xffffu) << 16);                   // tile_dim0 = n8
    g1[4] = 1;                                             // tile_dim1 = 1
    g1[5] = (int)n8;                                       // tensor_dim0_stride[31:0]
    g1[6] = 0;
    g1[7] = 0;
    i32x4 gz = {0, 0, 0, 0};
#if defined(__clang_major__) && __clang_major__ >= 23
    i32x8 gz8 = {0, 0, 0, 0, 0, 0, 0, 0};
    __builtin_amdgcn_tensor_load_to_lds(g0, g1, gz, gz, gz8, 0);
#else
    __builtin_amdgcn_tensor_load_to_lds(g0, g1, gz, gz, 0);
#endif
}
#endif

// ---------------------------------------------------------------------------
// WMMA GEMM:  C(M x NT*16) = A(M x KK) * Bt(NT*16 x KK, pre-transposed),
// bf16 in, f32 out. One 256-thread block = 8 waves; each wave owns a 16-row
// strip and all NT column tiles. Bt is already column-major in GLOBAL memory,
// so the LDS staging is a contiguous bulk copy -> Tensor Data Mover.
// A-fragment layout per CDNA5 ISA (16-bit A 16x32): lanes 0-15 -> K {0..7,
// 16..23}, lanes 16-31 -> K {8..15, 24..31}, row = lane&15.
// ---------------------------------------------------------------------------
template <int NT, int KK>
__global__ void __launch_bounds__(256)
wmma_gemm_bf16(const __bf16* __restrict__ A, int lda,
               const __bf16* __restrict__ Bg,     // NT*16*KK bf16, [col*KK + k]
               float* __restrict__ C, int ldc, int M)
{
    __shared__ __bf16 Bt[NT * 16 * KK];
#ifdef USE_TDM
    if (threadIdx.x < 32) {                 // wave 0 drives the DMA
        tdm_load_lds(Bg, Bt, (unsigned)(NT * 16 * KK / 4));  // n in 8B units
        __builtin_amdgcn_s_wait_tensorcnt(0);
    }
    __syncthreads();
#else
    for (int i = threadIdx.x; i < NT * 16 * KK; i += 256) Bt[i] = Bg[i];
    __syncthreads();
#endif

    const int tid     = threadIdx.x;
    const int wave    = tid >> 5;
    const int lane    = tid & 31;
    const int rowBase = blockIdx.x * 128 + wave * 16;
    if (rowBase >= M) return;

    const int hiHalf = (lane >= 16) ? 1 : 0;
    const int rA     = rowBase + (lane & 15);
    const int kOffA  = hiHalf ? 8 : 0;
    const int colN   = lane & 15;
    const int kOffB  = hiHalf ? 8 : 0;

    const v8f vzero = {0.f, 0.f, 0.f, 0.f, 0.f, 0.f, 0.f, 0.f};
    v8f acc[NT];
#pragma unroll
    for (int ct = 0; ct < NT; ++ct) acc[ct] = vzero;

#pragma unroll
    for (int kk = 0; kk < KK; kk += 32) {
        const __bf16* ap = A + (size_t)rA * lda + kk + kOffA;
        v8bf alo = *(const v8bf*)(ap);
        v8bf ahi = *(const v8bf*)(ap + 16);
        v16bf afrag = __builtin_shufflevector(alo, ahi,
            0, 1, 2, 3, 4, 5, 6, 7, 8, 9, 10, 11, 12, 13, 14, 15);
#pragma unroll
        for (int ct = 0; ct < NT; ++ct) {
            const __bf16* bp = &Bt[(ct * 16 + colN) * KK + kk + kOffB];
            v8bf blo = *(const v8bf*)(bp);
            v8bf bhi = *(const v8bf*)(bp + 16);
            v16bf bfrag = __builtin_shufflevector(blo, bhi,
                0, 1, 2, 3, 4, 5, 6, 7, 8, 9, 10, 11, 12, 13, 14, 15);
            acc[ct] = __builtin_amdgcn_wmma_f32_16x16x32_bf16(
                false, afrag, false, bfrag, (short)0, acc[ct], false, false);
        }
    }

    // C/D layout: VGPR j -> row rowBase + j + 8*hiHalf, col = tile*16 + (lane&15)
    const int rowAdj = rowBase + (hiHalf ? 8 : 0);
#pragma unroll
    for (int ct = 0; ct < NT; ++ct)
#pragma unroll
        for (int j = 0; j < 8; ++j)
            C[(size_t)(rowAdj + j) * ldc + ct * 16 + colN] = acc[ct][j];
}

// ---------------------------------------------------------------------------
// Small helper kernels
// ---------------------------------------------------------------------------
__global__ void cvt_bf16(const float* __restrict__ in, __bf16* __restrict__ outp, int n)
{
    int i = blockIdx.x * blockDim.x + threadIdx.x;
    if (i < n) outp[i] = f2bf(in[i]);
}

// Batched convert + transpose: in  = nb blocks of (K x C) row-major f32,
//                              out = nb blocks of (C x K) "column-major" bf16
__global__ void cvt_tr_bf16(const float* __restrict__ in, __bf16* __restrict__ outp,
                            int K, int Cc, int nb)
{
    int i = blockIdx.x * blockDim.x + threadIdx.x;
    int per = K * Cc;
    if (i >= nb * per) return;
    int b = i / per, rem = i - b * per;
    int k = rem / Cc, c = rem - k * Cc;
    outp[(size_t)b * per + (size_t)c * K + k] = f2bf(in[i]);
}

__global__ void init_feat(const float* __restrict__ x, float* __restrict__ cur)
{
    int i = blockIdx.x * blockDim.x + threadIdx.x;
    const size_t NF = (size_t)NN * HD;
    if (i < (int)NF) { float v = x[i]; cur[i] = v; cur[NF + i] = v; }
}

__global__ void copy_emb(const float* __restrict__ e0, float* __restrict__ emb)
{
    int i = threadIdx.x;
    if (i < RR * DD) emb[i] = e0[i];
}

// ra = rel_emb[r] @ W_rel[l,r]   -> (H*2*HID) = 256
__global__ void ra_kernel(const float* __restrict__ emb, const float* __restrict__ W_rel,
                          int l, int r, float* __restrict__ ra)
{
    int j = threadIdx.x;               // 256 threads
    float acc = 0.f;
    const float* e = emb + r * DD;
    const float* w = W_rel + (size_t)((l * RR + r) * DD) * (HH * 2 * HIDD);
    for (int k = 0; k < DD; ++k) acc += e[k] * w[(size_t)k * (HH * 2 * HIDD) + j];
    ra[j] = acc;
}

// rel_emb_next = rel_emb @ W_prop[l] + b_prop[l]  (both relations)
__global__ void emb_update(const float* __restrict__ emb, const float* __restrict__ W_prop,
                           const float* __restrict__ b_prop, int l, float* __restrict__ embn)
{
    int idx = threadIdx.x;             // 256 = R*128
    if (idx >= RR * DD) return;
    int r = idx >> 7, j = idx & 127;
    const float* w = W_prop + (size_t)((l * RR + r) * DD) * (HH * RHIDD);
    float acc = b_prop[(l * RR + r) * (HH * RHIDD) + j];
    for (int k = 0; k < DD; ++k) acc += emb[r * DD + k] * w[(size_t)k * (HH * RHIDD) + j];
    embn[idx] = acc;
}

// e_dst[n,h] = feat[n,h,:]·ra[h,:HID], e_src[n,h] = feat[n,h,:]·ra[h,HID:]
__global__ void node_attn(const float* __restrict__ feat, const float* __restrict__ ra,
                          float* __restrict__ esrc, float* __restrict__ edst)
{
    int idx = blockIdx.x * blockDim.x + threadIdx.x;
    if (idx >= NN * HH) return;
    int n = idx >> 2, h = idx & 3;
    const float* f = feat + (size_t)n * HD + h * HIDD;
    const float* rd = ra + h * (2 * HIDD);
    float a = 0.f, b = 0.f;
#pragma unroll
    for (int i = 0; i < HIDD; ++i) { a += f[i] * rd[i]; b += f[i] * rd[HIDD + i]; }
    edst[idx] = a;
    esrc[idx] = b;
}

// Pass 1: segment max over dst (encoded-uint atomicMax)
__global__ void edge_pass1(const int* __restrict__ src, const int* __restrict__ dst,
                           const float* __restrict__ esrc, const float* __restrict__ edst,
                           unsigned* __restrict__ menc, int E)
{
    int idx = blockIdx.x * blockDim.x + threadIdx.x;
    if (idx >= E * HH) return;
    int e = idx >> 2, h = idx & 3;
    int s = src[e], d = dst[e];
    float v = lrelu(esrc[s * HH + h] + edst[d * HH + h]);
    atomicMax(&menc[d * HH + h], fenc(v));
}

// Pass 2: ex = exp(e - m[d]); store per-edge; segment sum via atomicAdd
__global__ void edge_pass2(const int* __restrict__ src, const int* __restrict__ dst,
                           const float* __restrict__ esrc, const float* __restrict__ edst,
                           const unsigned* __restrict__ menc, float* __restrict__ aE,
                           float* __restrict__ ssum, int E)
{
    int idx = blockIdx.x * blockDim.x + threadIdx.x;
    if (idx >= E * HH) return;
    int e = idx >> 2, h = idx & 3;
    int s = src[e], d = dst[e];
    float v = lrelu(esrc[s * HH + h] + edst[d * HH + h]);
    float ex = __expf(v - fdec(menc[d * HH + h]));
    aE[e * HH + h] = ex;
    atomicAdd(&ssum[d * HH + h], ex);
}

// Pass 3: msg scatter; one wave (32 lanes == HID) per edge, loop heads.
__global__ void edge_scatter(const int* __restrict__ src, const int* __restrict__ dst,
                             const float* __restrict__ feat, const float* __restrict__ aE,
                             const float* __restrict__ ssum, float* __restrict__ msgout, int E)
{
    int e = blockIdx.x * 8 + (threadIdx.x >> 5);
    int lane = threadIdx.x & 31;
    if (e >= E) return;
    int s = src[e], d = dst[e];
    const float* fr = feat + (size_t)s * HD;
    __builtin_prefetch(fr, 0, 1);          // global_prefetch_b8
    float* op = msgout + (size_t)d * HD;
#pragma unroll
    for (int h = 0; h < HH; ++h) {
        float w = aE[e * HH + h] / ssum[d * HH + h];
        atomicAdd(&op[h * HIDD + lane], fr[h * HIDD + lane] * w);
    }
}

// out = relu(msg)*alpha + (res + b_res)*(1-alpha)
__global__ void blend(const float* __restrict__ msg, const float* __restrict__ res,
                      const float* __restrict__ b_res, const float* __restrict__ res_w,
                      int l, float* __restrict__ F4r)
{
    int i = blockIdx.x * blockDim.x + threadIdx.x;
    if (i >= NN * HD) return;
    int j = i & 127;
    float alpha = 1.f / (1.f + __expf(-res_w[l]));
    F4r[i] = fmaxf(msg[i], 0.f) * alpha + (res[i] + b_res[l * HD + j]) * (1.f - alpha);
}

// RelationCrossing: softmax over relations of lrelu(F4·A_cross), mix.
__global__ void relation_crossing(const float* __restrict__ F4, const float* __restrict__ A_cross,
                                  int l, float* __restrict__ nxt)
{
    int idx = blockIdx.x * blockDim.x + threadIdx.x;
    if (idx >= NN * HH) return;
    int n = idx >> 2, h = idx & 3;
    const size_t NF = (size_t)NN * HD;
    const float* f0 = F4 + (size_t)n * HD + h * HIDD;
    const float* f1 = f0 + NF;
    float v0[HIDD], v1[HIDD];
#pragma unroll
    for (int i = 0; i < HIDD; ++i) { v0[i] = f0[i]; v1[i] = f1[i]; }
#pragma unroll
    for (int ro = 0; ro < RR; ++ro) {
        const float* ac = A_cross + (size_t)((l * RR + ro) * HH + h) * HIDD;
        float d0 = 0.f, d1 = 0.f;
#pragma unroll
        for (int i = 0; i < HIDD; ++i) { d0 += v0[i] * ac[i]; d1 += v1[i] * ac[i]; }
        d0 = lrelu(d0); d1 = lrelu(d1);
        float m = fmaxf(d0, d1);
        float w0 = __expf(d0 - m), w1 = __expf(d1 - m);
        float inv = 1.f / (w0 + w1);
        w0 *= inv; w1 *= inv;
        float* op = nxt + (size_t)ro * NF + (size_t)n * HD + h * HIDD;
#pragma unroll
        for (int i = 0; i < HIDD; ++i) op[i] = w0 * v0[i] + w1 * v1[i];
    }
}

// rn[r,h,:] = rel_emb[r,h,:] @ W_fuse_rel[r,h]
__global__ void rn_kernel(const float* __restrict__ emb, const float* __restrict__ W_fuse_rel,
                          float* __restrict__ rnb)
{
    int idx = threadIdx.x;             // 256 = R*H*HID
    if (idx >= RR * HH * HIDD) return;
    int r = idx >> 7, rem = idx & 127, h = rem >> 5, d = rem & 31;
    float acc = 0.f;
    const float* w = W_fuse_rel + (size_t)((r * HH + h) * RHIDD) * HIDD;
    for (int c = 0; c < RHIDD; ++c) acc += emb[r * DD + h * RHIDD + c] * w[c * HIDD + d];
    rnb[idx] = acc;
}

// Final fusing: scores = softmax_r(lrelu(fn·rn)); out = sum_r fn*score
__global__ void relation_fusing(const float* __restrict__ fn, const float* __restrict__ rnb,
                                float* __restrict__ outp)
{
    int idx = blockIdx.x * blockDim.x + threadIdx.x;
    if (idx >= NN * HH) return;
    int n = idx >> 2, h = idx & 3;
    const size_t NF = (size_t)NN * HD;
    const float* f0 = fn + (size_t)n * HD + h * HIDD;
    const float* f1 = f0 + NF;
    const float* r0 = rnb + h * HIDD;
    const float* r1 = rnb + HH * HIDD + h * HIDD;
    float d0 = 0.f, d1 = 0.f;
#pragma unroll
    for (int i = 0; i < HIDD; ++i) { d0 += f0[i] * r0[i]; d1 += f1[i] * r1[i]; }
    d0 = lrelu(d0); d1 = lrelu(d1);
    float m = fmaxf(d0, d1);
    float w0 = __expf(d0 - m), w1 = __expf(d1 - m);
    float inv = 1.f / (w0 + w1);
    w0 *= inv; w1 *= inv;
    float* op = outp + (size_t)n * HD + h * HIDD;
#pragma unroll
    for (int i = 0; i < HIDD; ++i) op[i] = w0 * f0[i] + w1 * f1[i];
}

// ---------------------------------------------------------------------------
extern "C" void kernel_launch(void* const* d_in, const int* in_sizes, int n_in,
                              void* d_out, int out_size, void* d_ws, size_t ws_size,
                              hipStream_t stream)
{
    (void)in_sizes; (void)n_in; (void)out_size; (void)ws_size;
    const float* x          = (const float*)d_in[0];
    const float* W_node     = (const float*)d_in[1];
    const float* W_rel      = (const float*)d_in[2];
    const float* W_prop     = (const float*)d_in[3];
    const float* b_prop     = (const float*)d_in[4];
    const float* W_res      = (const float*)d_in[5];
    const float* b_res      = (const float*)d_in[6];
    const float* res_w      = (const float*)d_in[7];
    const float* A_cross    = (const float*)d_in[8];
    const float* rel_emb0   = (const float*)d_in[9];
    const float* W_fuse_node= (const float*)d_in[10];
    const float* W_fuse_rel = (const float*)d_in[11];
    const int*   src_idx    = (const int*)d_in[12];
    const int*   dst_idx    = (const int*)d_in[13];
    float* out = (float*)d_out;

    // ---- workspace carve ----
    char* base = (char*)d_ws;
    size_t off = 0;
    auto carve = [&](size_t bytes) -> void* {
        void* q = base + off;
        off += (bytes + 255) & ~(size_t)255;
        return q;
    };
    const size_t NF = (size_t)NN * HD;
    float*    bufA   = (float*)carve(RR * NF * 4);
    float*    bufB   = (float*)carve(RR * NF * 4);
    float*    F4     = (float*)carve(RR * NF * 4);    // conv_out, reused as fn_
    float*    featb  = (float*)carve(NF * 4);
    float*    resb   = (float*)carve(NF * 4);
    float*    msgout = (float*)carve(NF * 4);
    __bf16*   Abf    = (__bf16*)carve(NF * 2);
    __bf16*   Wbf    = (__bf16*)carve((size_t)DD * HD * 2);          // transposed weight
    __bf16*   Wfb    = (__bf16*)carve((size_t)RR * HH * HIDD * HIDD * 2);
    float*    esrc   = (float*)carve((size_t)NN * HH * 4);
    float*    edstb  = (float*)carve((size_t)NN * HH * 4);
    unsigned* menc   = (unsigned*)carve((size_t)NN * HH * 4);
    float*    ssum   = (float*)carve((size_t)NN * HH * 4);
    float*    aE     = (float*)carve((size_t)EE * HH * 4);
    float*    rab    = (float*)carve(256 * 4);
    float*    embA   = (float*)carve(RR * DD * 4);
    float*    embB   = (float*)carve(RR * DD * 4);
    float*    rnb    = (float*)carve(RR * HH * HIDD * 4);

    const int T    = 256;
    const int gNM  = (int)((NF + T - 1) / T);
    const int gNH  = (NN * HH + T - 1) / T;
    const int gE4  = (EE * HH + T - 1) / T;
    const int gE8  = (EE + 7) / 8;
    const int gG   = (NN + 127) / 128;
    const int gW   = (DD * HD + T - 1) / T;

    init_feat<<<gNM, T, 0, stream>>>(x, bufA);
    copy_emb<<<1, T, 0, stream>>>(rel_emb0, embA);

    float* cur = bufA; float* nxt = bufB;
    float* embc = embA; float* embn = embB;

    for (int l = 0; l < LL; ++l) {
        for (int r = 0; r < RR; ++r) {
            float* curR = cur + (size_t)r * NF;
            const int* sr = src_idx + (size_t)r * EE;
            const int* dr = dst_idx + (size_t)r * EE;

            ra_kernel<<<1, T, 0, stream>>>(embc, W_rel, l, r, rab);

            // feat = cur[r] @ W_node[l]  (bf16 WMMA, f32 accumulate)
            cvt_bf16<<<gNM, T, 0, stream>>>(curR, Abf, (int)NF);
            cvt_tr_bf16<<<gW, T, 0, stream>>>(W_node + (size_t)l * DD * HD, Wbf, DD, HD, 1);
            wmma_gemm_bf16<8, 128><<<gG, T, 0, stream>>>(Abf, HD, Wbf, featb, HD, NN);

            node_attn<<<gNH, T, 0, stream>>>(featb, rab, esrc, edstb);

            hipMemsetAsync(menc, 0, (size_t)NN * HH * 4, stream);
            hipMemsetAsync(ssum, 0, (size_t)NN * HH * 4, stream);
            hipMemsetAsync(msgout, 0, NF * 4, stream);

            edge_pass1<<<gE4, T, 0, stream>>>(sr, dr, esrc, edstb, menc, EE);
            edge_pass2<<<gE4, T, 0, stream>>>(sr, dr, esrc, edstb, menc, aE, ssum, EE);
            edge_scatter<<<gE8, T, 0, stream>>>(sr, dr, featb, aE, ssum, msgout, EE);

            // residual: res = cur[r] @ W_res[l]
            cvt_tr_bf16<<<gW, T, 0, stream>>>(W_res + (size_t)l * DD * HD, Wbf, DD, HD, 1);
            wmma_gemm_bf16<8, 128><<<gG, T, 0, stream>>>(Abf, HD, Wbf, resb, HD, NN);

            blend<<<gNM, T, 0, stream>>>(msgout, resb, b_res, res_w, l, F4 + (size_t)r * NF);
        }
        relation_crossing<<<gNH, T, 0, stream>>>(F4, A_cross, l, nxt);
        emb_update<<<1, T, 0, stream>>>(embc, W_prop, b_prop, l, embn);
        { float* t = cur; cur = nxt; nxt = t; }
        { float* t = embc; embc = embn; embn = t; }
    }

    // ---- final relation fusing ----
    cvt_tr_bf16<<<(RR * HH * HIDD * HIDD + T - 1) / T, T, 0, stream>>>(
        W_fuse_node, Wfb, HIDD, HIDD, RR * HH);
    rn_kernel<<<1, T, 0, stream>>>(embc, W_fuse_rel, rnb);

    float* fn = F4;   // F4 is free after the last crossing; reuse as fn_
    for (int r = 0; r < RR; ++r) {
        cvt_bf16<<<gNM, T, 0, stream>>>(cur + (size_t)r * NF, Abf, (int)NF);
        for (int h = 0; h < HH; ++h) {
            wmma_gemm_bf16<2, 32><<<gG, T, 0, stream>>>(
                Abf + h * HIDD, HD,
                Wfb + (size_t)(r * HH + h) * HIDD * HIDD,
                fn + (size_t)r * NF + h * HIDD, HD, NN);
        }
    }
    relation_fusing<<<gNH, T, 0, stream>>>(fn, rnb, out);
}
